// ProgramCloneModel_42743514530123
// MI455X (gfx1250) — matrix operand order repaired and checked by reading, not compiled, not run on hardware
//
#include <hip/hip_runtime.h>
#include <hip/hip_bf16.h>

// ---------------------------------------------------------------------------
// CDNA5 (gfx1250) relational-GNN pipeline.
// All GEMMs use v_wmma_f32_16x16x32_f16 (wave32), fp32 accumulation.
// ---------------------------------------------------------------------------

typedef __attribute__((ext_vector_type(16))) _Float16 v16h;
typedef __attribute__((ext_vector_type(8)))  float    v8f;

#define HID     256
#define ETD     32
#define INDIM   128
#define MSG_K   (HID + ETD)   // 288
#define LN_EPS  1e-5f

union Frag16 { v16h v; uint4 q[2]; };

// ---------------------------------------------------------------------------
// Small utility kernels
// ---------------------------------------------------------------------------
__global__ void k_zero(float* p, int n) {
    int i = blockIdx.x * blockDim.x + threadIdx.x;
    if (i < n) p[i] = 0.0f;
}

__global__ void k_deg(const int* __restrict__ dst, int E, float* __restrict__ deg) {
    int i = blockIdx.x * blockDim.x + threadIdx.x;
    if (i < E) atomicAdd(&deg[dst[i]], 1.0f);
}

__global__ void k_invdeg(float* deg, int n) {   // in-place deg -> 1/max(deg,1)
    int i = blockIdx.x * blockDim.x + threadIdx.x;
    if (i < n) deg[i] = 1.0f / fmaxf(deg[i], 1.0f);
}

__global__ void k_f32_to_f16(const float* __restrict__ s, _Float16* __restrict__ d, int n) {
    int i = blockIdx.x * blockDim.x + threadIdx.x;
    if (i < n) d[i] = (_Float16)s[i];
}

// out16[i] = f16( agg[i] * invdeg[row] ), C = 256 columns
__global__ void k_scale_rows_f16(const float* __restrict__ agg, const float* __restrict__ invdeg,
                                 _Float16* __restrict__ out, int n) {
    int i = blockIdx.x * blockDim.x + threadIdx.x;
    if (i < n) out[i] = (_Float16)(agg[i] * invdeg[i >> 8]);
}

// Pack W[N(out) x K(in)] fp32 row-major -> f16 WMMA B-operand layout:
// dst[(((kc*16 + nt)*32 + lane)<<4) + v], lane = khalf*16 + (n&15),
// K of element v = kc*32 + khalf*16 + v  (contiguous per lane -> 32B loads).
__global__ void k_pack_w(const float* __restrict__ W, _Float16* __restrict__ dst, int N, int K) {
    int i = blockIdx.x * blockDim.x + threadIdx.x;
    if (i >= N * K) return;
    int n = i / K, k = i - n * K;
    int kc = k >> 5, kr = k & 31, khalf = kr >> 4, v = kr & 15;
    int nt = n >> 4, nl = n & 15;
    int lane = khalf * 16 + nl;
    int NT = N >> 4;
    dst[(((kc * NT + nt) * 32 + lane) << 4) + v] = (_Float16)W[i];
}

// ---------------------------------------------------------------------------
// Edge message GEMM + ReLU + atomic scatter (segment_sum)
//   msg = relu( concat(h[src], ef[etype]) @ Wmsg^T + b );  agg[dst] += msg
// Block = 256 threads (8 waves), 64 edges. Wave(w): Mtile = w>>1, N-half = w&1.
// ---------------------------------------------------------------------------
__global__ void __launch_bounds__(256) k_msg(
    const _Float16* __restrict__ hf,     // [N][256] f16
    const int* __restrict__ src, const int* __restrict__ dst, const int* __restrict__ et,
    const _Float16* __restrict__ ef,     // [16][32] f16 (layer slice)
    const _Float16* __restrict__ Bp,     // packed [9][16][32][16] f16
    const float* __restrict__ bias,      // [256] fp32 (layer slice)
    float* __restrict__ agg, int E)
{
    __shared__ _Float16 ldsA[64][MSG_K];   // 36 KB
    __shared__ int s_src[64], s_dst[64], s_et[64];

    const int tid   = threadIdx.x;
    const int ebase = blockIdx.x * 64;

    if (tid < 64) {
        int e = ebase + tid;
        if (e < E) { s_src[tid] = src[e]; s_dst[tid] = dst[e]; s_et[tid] = et[e]; }
        else       { s_src[tid] = 0;      s_dst[tid] = -1;     s_et[tid] = 0;     }
    }
    __syncthreads();

    // Stage A = [h[src] | ef[etype]] : 64 rows x 144 dwords (288 f16)
    const uint* hfd = (const uint*)hf;
    const uint* efd = (const uint*)ef;
    for (int i = tid; i < 64 * 144; i += 256) {
        int e = i / 144, d = i - e * 144;
        uint val = 0u;
        if (ebase + e < E)
            val = (d < 128) ? hfd[s_src[e] * 128 + d] : efd[s_et[e] * 16 + (d - 128)];
        ((uint*)&ldsA[e][0])[d] = val;
    }
    __syncthreads();

    const int w = tid >> 5, lane = tid & 31;
    const int mt = w >> 1, nh = w & 1;
    const int row = mt * 16 + (lane & 15), half = lane >> 4;

    const v8f vzero = {0.f, 0.f, 0.f, 0.f, 0.f, 0.f, 0.f, 0.f};
    v8f acc[8];
    #pragma unroll
    for (int i = 0; i < 8; ++i) acc[i] = vzero;

    #pragma unroll
    for (int kc = 0; kc < 9; ++kc) {
        Frag16 a;
        a.q[0] = *(const uint4*)&ldsA[row][kc * 32 + half * 16];
        a.q[1] = *(const uint4*)&ldsA[row][kc * 32 + half * 16 + 8];
        #pragma unroll
        for (int nt = 0; nt < 8; ++nt) {
            int ntg = nh * 8 + nt;
            v16h b = *(const v16h*)(Bp + (((kc * 16 + ntg) * 32 + lane) << 4));
            acc[nt] = __builtin_amdgcn_wmma_f32_16x16x32_f16(
                false, a.v, false, b, (short)0, acc[nt], false, false);
        }
    }

    // Epilogue: bias + ReLU + scatter-add.  C layout: VGPR r, lane l ->
    // row M = (l>>4)*8 + r, col N = l&15.
    const int cl = lane & 15, rh = lane >> 4;
    #pragma unroll
    for (int nt = 0; nt < 8; ++nt) {
        int col = (nh * 8 + nt) * 16 + cl;
        float bv = bias[col];
        #pragma unroll
        for (int r = 0; r < 8; ++r) {
            int eloc = mt * 16 + rh * 8 + r;
            int dn = s_dst[eloc];
            if (dn >= 0) {
                float v = acc[nt][r] + bv;
                v = v > 0.f ? v : 0.f;
                atomicAdd(&agg[dn * 256 + col], v);
            }
        }
    }
}

// ---------------------------------------------------------------------------
// Dense GEMM: out[M x 256] = A[M x K] @ W^T (+bias) [opt relu, opt f16 copy]
// Same WMMA skeleton; K in {128, 256}, ksh = log2(K/2).
// ---------------------------------------------------------------------------
__global__ void __launch_bounds__(256) k_gemm(
    const _Float16* __restrict__ Af,   // [M][K] f16
    const _Float16* __restrict__ Bp,   // packed [(K/32)][16][32][16]
    const float* __restrict__ bias,    // [256]
    float* __restrict__ out,           // [M][256]
    _Float16* __restrict__ out16,      // optional (may be null)
    int M, int K, int ksh, int relu)
{
    __shared__ _Float16 ldsA[64][HID];  // up to 32 KB

    const int tid = threadIdx.x;
    const int rbase = blockIdx.x * 64;
    const int kdw = 1 << ksh;           // dwords per row
    const uint* Ad = (const uint*)Af;

    for (int i = tid; i < (64 << ksh); i += 256) {
        int e = i >> ksh, d = i & (kdw - 1);
        int gr = rbase + e;
        uint val = (gr < M) ? Ad[gr * kdw + d] : 0u;
        ((uint*)&ldsA[e][0])[d] = val;
    }
    __syncthreads();

    const int w = tid >> 5, lane = tid & 31;
    const int mt = w >> 1, nh = w & 1;
    const int row = mt * 16 + (lane & 15), half = lane >> 4;
    const int KC = K >> 5;

    const v8f vzero = {0.f, 0.f, 0.f, 0.f, 0.f, 0.f, 0.f, 0.f};
    v8f acc[8];
    #pragma unroll
    for (int i = 0; i < 8; ++i) acc[i] = vzero;

    for (int kc = 0; kc < KC; ++kc) {
        Frag16 a;
        a.q[0] = *(const uint4*)&ldsA[row][kc * 32 + half * 16];
        a.q[1] = *(const uint4*)&ldsA[row][kc * 32 + half * 16 + 8];
        #pragma unroll
        for (int nt = 0; nt < 8; ++nt) {
            int ntg = nh * 8 + nt;
            v16h b = *(const v16h*)(Bp + (((kc * 16 + ntg) * 32 + lane) << 4));
            acc[nt] = __builtin_amdgcn_wmma_f32_16x16x32_f16(
                false, a.v, false, b, (short)0, acc[nt], false, false);
        }
    }

    const int cl = lane & 15, rh = lane >> 4;
    #pragma unroll
    for (int nt = 0; nt < 8; ++nt) {
        int col = (nh * 8 + nt) * 16 + cl;
        float bv = bias[col];
        #pragma unroll
        for (int r = 0; r < 8; ++r) {
            int gr = rbase + mt * 16 + rh * 8 + r;
            if (gr < M) {
                float v = acc[nt][r] + bv;
                if (relu) v = fmaxf(v, 0.f);
                out[gr * 256 + col] = v;
                if (out16) out16[gr * 256 + col] = (_Float16)v;
            }
        }
    }
}

// ---------------------------------------------------------------------------
// LayerNorm over residual: h = LN(h + upd) * g + b ; also write f16 copy.
// One wave (32 lanes) per row; 8 rows per 256-thread block.
// ---------------------------------------------------------------------------
__global__ void __launch_bounds__(256) k_ln(
    float* __restrict__ h, const float* __restrict__ upd,
    const float* __restrict__ g, const float* __restrict__ b,
    _Float16* __restrict__ hf, int N)
{
    const int w = threadIdx.x >> 5, lane = threadIdx.x & 31;
    const int row = blockIdx.x * 8 + w;
    if (row >= N) return;

    float z[8];
    float s = 0.f;
    #pragma unroll
    for (int j = 0; j < 8; ++j) {
        int c = lane + j * 32;
        z[j] = h[row * 256 + c] + upd[row * 256 + c];
        s += z[j];
    }
    #pragma unroll
    for (int o = 16; o; o >>= 1) s += __shfl_xor(s, o, 32);
    const float mu = s * (1.0f / 256.0f);

    float vs = 0.f;
    #pragma unroll
    for (int j = 0; j < 8; ++j) { float d = z[j] - mu; vs += d * d; }
    #pragma unroll
    for (int o = 16; o; o >>= 1) vs += __shfl_xor(vs, o, 32);
    const float inv = rsqrtf(vs * (1.0f / 256.0f) + LN_EPS);

    #pragma unroll
    for (int j = 0; j < 8; ++j) {
        int c = lane + j * 32;
        float v = g[c] * (z[j] - mu) * inv + b[c];
        h[row * 256 + c]  = v;
        hf[row * 256 + c] = (_Float16)v;
    }
}

// Column mean over nodes: out[c] = mean_n h[n][c]; one block per column.
__global__ void __launch_bounds__(256) k_colmean(const float* __restrict__ h, int N,
                                                 float* __restrict__ out) {
    const int c = blockIdx.x;
    float s = 0.f;
    for (int n = threadIdx.x; n < N; n += 256) s += h[n * 256 + c];
    __shared__ float red[256];
    red[threadIdx.x] = s;
    __syncthreads();
    for (int st = 128; st; st >>= 1) {
        if (threadIdx.x < st) red[threadIdx.x] += red[threadIdx.x + st];
        __syncthreads();
    }
    if (threadIdx.x == 0) out[c] = red[0] / (float)N;
}

// Pair MLP head (tiny, fp32, single block).
__global__ void __launch_bounds__(256) k_head(
    const float* __restrict__ ea, const float* __restrict__ eb,
    const float* __restrict__ W1, const float* __restrict__ b1,
    const float* __restrict__ W2, const float* __restrict__ b2,
    const float* __restrict__ W3, const float* __restrict__ b3,
    float* __restrict__ logit)
{
    __shared__ float feat[1024];
    __shared__ float h1[512];
    __shared__ float h2[256];
    __shared__ float red[256];
    const int t = threadIdx.x;

    float a = ea[t], b = eb[t];
    feat[t]       = a;
    feat[256 + t] = b;
    feat[512 + t] = fabsf(a - b);
    feat[768 + t] = a * b;
    __syncthreads();

    for (int o = t; o < 512; o += 256) {
        float s = b1[o];
        for (int k = 0; k < 1024; ++k) s += W1[o * 1024 + k] * feat[k];
        h1[o] = fmaxf(s, 0.f);
    }
    __syncthreads();

    {
        float s = b2[t];
        for (int k = 0; k < 512; ++k) s += W2[t * 512 + k] * h1[k];
        h2[t] = fmaxf(s, 0.f);
    }
    __syncthreads();

    red[t] = h2[t] * W3[t];
    __syncthreads();
    for (int st = 128; st; st >>= 1) {
        if (t < st) red[t] += red[t + st];
        __syncthreads();
    }
    if (t == 0) logit[0] = red[0] + b3[0];
}

// ---------------------------------------------------------------------------
// Host orchestration
// ---------------------------------------------------------------------------
extern "C" void kernel_launch(void* const* d_in, const int* in_sizes, int n_in,
                              void* d_out, int out_size, void* d_ws, size_t ws_size,
                              hipStream_t stream)
{
    (void)n_in; (void)out_size; (void)ws_size;

    const float* x_a      = (const float*)d_in[0];
    const int*   ei_a     = (const int*)  d_in[1];
    const int*   et_a     = (const int*)  d_in[2];
    const float* x_b      = (const float*)d_in[3];
    const int*   ei_b     = (const int*)  d_in[4];
    const int*   et_b     = (const int*)  d_in[5];
    const float* W_in     = (const float*)d_in[6];
    const float* b_in     = (const float*)d_in[7];
    const float* eembs    = (const float*)d_in[8];
    const float* W_msg    = (const float*)d_in[9];
    const float* b_msg    = (const float*)d_in[10];
    const float* W_upd    = (const float*)d_in[11];
    const float* b_upd    = (const float*)d_in[12];
    const float* ln_g     = (const float*)d_in[13];
    const float* ln_b     = (const float*)d_in[14];
    const float* W1       = (const float*)d_in[15];
    const float* b1       = (const float*)d_in[16];
    const float* W2       = (const float*)d_in[17];
    const float* b2       = (const float*)d_in[18];
    const float* W3       = (const float*)d_in[19];
    const float* b3       = (const float*)d_in[20];

    const int N = in_sizes[0] / INDIM;   // 20000
    const int E = in_sizes[1] / 2;       // 320000
    const int L = 3;

    // Carve workspace
    char* p = (char*)d_ws;
    auto carve = [&](size_t bytes) -> void* {
        void* r = (void*)p;
        p += (bytes + 255) & ~(size_t)255;
        return r;
    };
    _Float16* Win_p  = (_Float16*)carve((size_t)256 * 128 * 2);
    _Float16* Wmsg_p = (_Float16*)carve((size_t)L * 256 * MSG_K * 2);
    _Float16* Wupd_p = (_Float16*)carve((size_t)L * 256 * 256 * 2);
    _Float16* ef16   = (_Float16*)carve((size_t)L * 16 * 32 * 2);
    _Float16* xf16   = (_Float16*)carve((size_t)N * INDIM * 2);
    _Float16* hf16   = (_Float16*)carve((size_t)N * 256 * 2);
    _Float16* agg16  = (_Float16*)carve((size_t)N * 256 * 2);
    float*    hbuf   = (float*)   carve((size_t)N * 256 * 4);
    float*    aggbuf = (float*)   carve((size_t)N * 256 * 4);
    float*    updbuf = (float*)   carve((size_t)N * 256 * 4);
    float*    degv   = (float*)   carve((size_t)N * 4);

    const int T = 256;
    auto blocks = [](int n, int per) { return (n + per - 1) / per; };

    // ---- Weight packing (deterministic, every call) ----
    k_pack_w<<<blocks(256 * 128, T), T, 0, stream>>>(W_in, Win_p, 256, 128);
    for (int l = 0; l < L; ++l) {
        k_pack_w<<<blocks(256 * MSG_K, T), T, 0, stream>>>(
            W_msg + (size_t)l * 256 * MSG_K, Wmsg_p + (size_t)l * 256 * MSG_K, 256, MSG_K);
        k_pack_w<<<blocks(256 * 256, T), T, 0, stream>>>(
            W_upd + (size_t)l * 256 * 256, Wupd_p + (size_t)l * 256 * 256, 256, 256);
    }
    k_f32_to_f16<<<blocks(L * 16 * 32, T), T, 0, stream>>>(eembs, ef16, L * 16 * 32);

    float* out   = (float*)d_out;
    float* emb_a = out + 1;
    float* emb_b = out + 1 + 256;

    for (int g = 0; g < 2; ++g) {
        const float* x   = g ? x_b  : x_a;
        const int*   ei  = g ? ei_b : ei_a;
        const int*   et  = g ? et_b : et_a;
        const int*   src = ei;
        const int*   dst = ei + E;
        float* emb = g ? emb_b : emb_a;

        // degree -> 1/max(deg,1)
        k_zero<<<blocks(N, T), T, 0, stream>>>(degv, N);
        k_deg<<<blocks(E, T), T, 0, stream>>>(dst, E, degv);
        k_invdeg<<<blocks(N, T), T, 0, stream>>>(degv, N);

        // h = relu(x @ Win^T + b_in)
        k_f32_to_f16<<<blocks(N * INDIM, T), T, 0, stream>>>(x, xf16, N * INDIM);
        k_gemm<<<blocks(N, 64), T, 0, stream>>>(xf16, Win_p, b_in, hbuf, hf16,
                                                N, 128, 6, 1);

        for (int l = 0; l < L; ++l) {
            k_zero<<<blocks(N * 256, T), T, 0, stream>>>(aggbuf, N * 256);
            k_msg<<<blocks(E, 64), T, 0, stream>>>(
                hf16, src, dst, et,
                ef16 + (size_t)l * 16 * 32,
                Wmsg_p + (size_t)l * 256 * MSG_K,
                b_msg + (size_t)l * 256,
                aggbuf, E);
            k_scale_rows_f16<<<blocks(N * 256, T), T, 0, stream>>>(aggbuf, degv, agg16, N * 256);
            k_gemm<<<blocks(N, 64), T, 0, stream>>>(
                agg16, Wupd_p + (size_t)l * 256 * 256, b_upd + (size_t)l * 256,
                updbuf, (_Float16*)nullptr, N, 256, 7, 0);
            k_ln<<<blocks(N, 8), T, 0, stream>>>(
                hbuf, updbuf, ln_g + (size_t)l * 256, ln_b + (size_t)l * 256, hf16, N);
        }

        k_colmean<<<256, T, 0, stream>>>(hbuf, N, emb);
    }

    k_head<<<1, T, 0, stream>>>(emb_a, emb_b, W1, b1, W2, b2, W3, b3, out);
}
